// Graph_GCN_21638045237568
// MI455X (gfx1250) — matrix operand order repaired and checked by reading, not compiled, not run on hardware
//
#include <hip/hip_runtime.h>
#include <math.h>

typedef __bf16 bf16;
typedef __attribute__((ext_vector_type(16))) __bf16 v16bf;
typedef __attribute__((ext_vector_type(8)))  __bf16 v8bf;
typedef __attribute__((ext_vector_type(8)))  float  v8f;
typedef __attribute__((ext_vector_type(4)))  float  v4f;

#define V_DIM 8192
#define B_DIM 64
#define KSPLIT 4
#define VB ((size_t)V_DIM * B_DIM)

// ---------------------------------------------------------------------------
// Weight transpose + fp32 -> bf16 convert:  Wt[n*K + k] = (bf16) W[k*N + n]
// LDS-tiled so both global read and global write are coalesced.
// ---------------------------------------------------------------------------
__global__ void transpose_w_bf16(const float* __restrict__ W,
                                 bf16* __restrict__ Wt, int K, int N) {
  __shared__ float tile[32][33];
  const int k0 = blockIdx.x * 32;
  const int n0 = blockIdx.y * 32;
  const int tx = threadIdx.x, ty = threadIdx.y;
#pragma unroll
  for (int r = 0; r < 32; r += 8)
    tile[ty + r][tx] = W[(size_t)(k0 + ty + r) * N + (n0 + tx)];
  __syncthreads();
#pragma unroll
  for (int r = 0; r < 32; r += 8)
    Wt[(size_t)(n0 + ty + r) * K + (k0 + tx)] = (bf16)tile[tx][ty + r];
}

// ---------------------------------------------------------------------------
// x_in [64][8192] fp32  ->  T0 [V][B] fp32 (= x^T)  and Tb0 [B][V] bf16
// ---------------------------------------------------------------------------
__global__ void prep_x(const float* __restrict__ x,
                       float* __restrict__ T0, bf16* __restrict__ Tb0) {
  const int i = blockIdx.x * 256 + threadIdx.x;   // 64*8192 threads
  const int b = i >> 13;
  const int v = i & (V_DIM - 1);
  const float val = x[i];
  Tb0[i] = (bf16)val;
  T0[(size_t)v * B_DIM + b] = val;
}

// ---------------------------------------------------------------------------
// Chebyshev GEMM, K-split partial: Ypart[kc] = L[:, kc-chunk] @ X[kc-chunk, :]
//   L     : [V][V] fp32 row-major (converted to bf16 in registers, NT loads)
//   Xt    : [B][V] bf16  (T_{k-1} transposed -> contiguous B-fragment loads)
//   Ypart : [KSPLIT][V][B] fp32 partial products
// One wave computes a 16(M) x 64(N) strip over K/KSPLIT; 2 waves per block.
// Grid: (V/32, KSPLIT) => 2048 waves for latency hiding at HBM roofline.
// ---------------------------------------------------------------------------
__launch_bounds__(64)
__global__ void cheb_gemm_part(const float* __restrict__ Lm,
                               const bf16* __restrict__ Xt,
                               float* __restrict__ Ypart) {
  const int wave = threadIdx.x >> 5;
  const int lane = threadIdx.x & 31;
  const int g    = lane >> 4;      // lane-half selects K sub-block
  const int ml   = lane & 15;      // M row (A) / N col (B)
  const int m0   = blockIdx.x * 32 + wave * 16;
  const int kBeg = blockIdx.y * (V_DIM / KSPLIT);
  const int kEnd = kBeg + (V_DIM / KSPLIT);
  const float* Arow = Lm + (size_t)(m0 + ml) * V_DIM;

  v8f acc0 = {}, acc1 = {}, acc2 = {}, acc3 = {};
  for (int k0 = kBeg; k0 < kEnd; k0 += 32) {
    // A fragment: rows of L, fp32 -> bf16 in registers. The
    // lane holds K = {g*16..g*16+15} (contiguous, matching B).
    const float* p = Arow + k0 + g * 16;
    v4f x0 = __builtin_nontemporal_load((const v4f*)(p));
    v4f x1 = __builtin_nontemporal_load((const v4f*)(p + 4));
    v4f x2 = __builtin_nontemporal_load((const v4f*)(p + 8));
    v4f x3 = __builtin_nontemporal_load((const v4f*)(p + 12));
    v16bf a;
#pragma unroll
    for (int i = 0; i < 4; ++i) {
      a[i]      = (bf16)x0[i];
      a[4 + i]  = (bf16)x1[i];
      a[8 + i]  = (bf16)x2[i];
      a[12 + i] = (bf16)x3[i];
    }
    // B fragments: lane holds column n = tile_base + ml, K = k0+g*16 .. +15
    const bf16* brow = Xt + k0 + g * 16;
    v16bf b0 = *(const v16bf*)(brow + (size_t)(ml)      * V_DIM);
    v16bf b1 = *(const v16bf*)(brow + (size_t)(16 + ml) * V_DIM);
    v16bf b2 = *(const v16bf*)(brow + (size_t)(32 + ml) * V_DIM);
    v16bf b3 = *(const v16bf*)(brow + (size_t)(48 + ml) * V_DIM);
    acc0 = __builtin_amdgcn_wmma_f32_16x16x32_bf16(false, a, false, b0, (short)0, acc0, false, false);
    acc1 = __builtin_amdgcn_wmma_f32_16x16x32_bf16(false, a, false, b1, (short)0, acc1, false, false);
    acc2 = __builtin_amdgcn_wmma_f32_16x16x32_bf16(false, a, false, b2, (short)0, acc2, false, false);
    acc3 = __builtin_amdgcn_wmma_f32_16x16x32_bf16(false, a, false, b3, (short)0, acc3, false, false);
  }
  float* Yp = Ypart + (size_t)blockIdx.y * VB;
  const v8f accs[4] = {acc0, acc1, acc2, acc3};
#pragma unroll
  for (int j = 0; j < 4; ++j) {
#pragma unroll
    for (int r = 0; r < 8; ++r) {
      const int row = m0 + r + 8 * g;   // vertex index
      const int col = j * 16 + ml;      // batch index
      Yp[(size_t)row * B_DIM + col] = accs[j][r];
    }
  }
}

// ---------------------------------------------------------------------------
// Combine K-split partials + Chebyshev recurrence:
//   T = alpha * sum_c Ypart[c]  -  Cprev   (Cprev may be null)
// Writes fp32 master [V][B] and bf16 transposed copy [B][V].
// ---------------------------------------------------------------------------
__global__ void cheb_combine(const float* __restrict__ Ypart,
                             const float* __restrict__ Cprev,
                             float alpha,
                             float* __restrict__ Tout,
                             bf16* __restrict__ Tbout) {
  const int i = blockIdx.x * 256 + threadIdx.x;   // over V*B, b fastest
  const int v = i >> 6;
  const int b = i & (B_DIM - 1);
  float y = Ypart[i] + Ypart[i + VB] + Ypart[i + 2 * VB] + Ypart[i + 3 * VB];
  y *= alpha;
  if (Cprev) y -= Cprev[i];
  Tout[i] = y;
  Tbout[(size_t)b * V_DIM + v] = (bf16)y;
}

// ---------------------------------------------------------------------------
// cl1 (Chebyshev feature mix, K=5 -> F=16) + ReLU + max-pool over 8 vertices.
// T: [5][V][B] fp32 ; xcp: [B][16384] bf16 with column = vblock*16 + f
// ReLU and max commute, so pool first then clamp once.
// ---------------------------------------------------------------------------
__global__ void cl1_pool(const float* __restrict__ T,
                         const float* __restrict__ cl1_W,  // [5][16]
                         const float* __restrict__ cl1_b,  // [16]
                         bf16* __restrict__ xcp) {
  const int b  = threadIdx.x;     // 64
  const int f  = threadIdx.y;     // 16
  const int vb = blockIdx.x;      // 1024 vertex blocks
  float w[5];
#pragma unroll
  for (int k = 0; k < 5; ++k) w[k] = cl1_W[k * 16 + f];
  const float bias = cl1_b[f];
  float best = -1e30f;
#pragma unroll
  for (int p = 0; p < 8; ++p) {
    const int v = vb * 8 + p;
    float s = bias;
#pragma unroll
    for (int k = 0; k < 5; ++k)
      s += T[(size_t)k * VB + (size_t)v * B_DIM + b] * w[k];
    best = fmaxf(best, s);
  }
  best = fmaxf(best, 0.f);
  xcp[(size_t)b * 16384 + vb * 16 + f] = (bf16)best;
}

// ---------------------------------------------------------------------------
// Generic FC GEMM: C[64,N] = act(A[64,K] @ Wt[N,K]^T + bias)
// A, Wt bf16 row-major; C fp32; optional bf16 mirror Cb (input to next layer).
// Block = 4 waves covering M=64; grid.x = N/64.
// ---------------------------------------------------------------------------
__launch_bounds__(128)
__global__ void fc_gemm(const bf16* __restrict__ A,
                        const bf16* __restrict__ Wt,
                        const float* __restrict__ bias,
                        float* __restrict__ C,
                        bf16* __restrict__ Cb,
                        int K, int N, int relu) {
  const int wave = threadIdx.x >> 5;
  const int lane = threadIdx.x & 31;
  const int g    = lane >> 4;
  const int ml   = lane & 15;
  const int m0   = wave * 16;
  const int n0   = blockIdx.x * 64;
  const bf16* Arow = A + (size_t)(m0 + ml) * K;

  v8f acc0 = {}, acc1 = {}, acc2 = {}, acc3 = {};
  for (int k0 = 0; k0 < K; k0 += 32) {
    v8bf lo = *(const v8bf*)(Arow + k0 + g * 16);
    v8bf hi = *(const v8bf*)(Arow + k0 + g * 16 + 8);
    v16bf a;
#pragma unroll
    for (int i = 0; i < 8; ++i) { a[i] = lo[i]; a[8 + i] = hi[i]; }
    const bf16* brow = Wt + k0 + g * 16;
    v16bf b0 = *(const v16bf*)(brow + (size_t)(n0 + ml)      * K);
    v16bf b1 = *(const v16bf*)(brow + (size_t)(n0 + 16 + ml) * K);
    v16bf b2 = *(const v16bf*)(brow + (size_t)(n0 + 32 + ml) * K);
    v16bf b3 = *(const v16bf*)(brow + (size_t)(n0 + 48 + ml) * K);
    acc0 = __builtin_amdgcn_wmma_f32_16x16x32_bf16(false, a, false, b0, (short)0, acc0, false, false);
    acc1 = __builtin_amdgcn_wmma_f32_16x16x32_bf16(false, a, false, b1, (short)0, acc1, false, false);
    acc2 = __builtin_amdgcn_wmma_f32_16x16x32_bf16(false, a, false, b2, (short)0, acc2, false, false);
    acc3 = __builtin_amdgcn_wmma_f32_16x16x32_bf16(false, a, false, b3, (short)0, acc3, false, false);
  }
  const v8f accs[4] = {acc0, acc1, acc2, acc3};
#pragma unroll
  for (int j = 0; j < 4; ++j) {
#pragma unroll
    for (int r = 0; r < 8; ++r) {
      const int m = m0 + r + 8 * g;
      const int n = n0 + j * 16 + ml;
      float y = accs[j][r] + bias[n];
      if (relu) y = fmaxf(y, 0.f);
      C[(size_t)m * N + n] = y;
      if (Cb) Cb[(size_t)m * N + n] = (bf16)y;
    }
  }
}

// ---------------------------------------------------------------------------
// z = [hidden | xn] @ sum2_W + sum2_b ; logp = log_softmax(z). One wave/row.
// ---------------------------------------------------------------------------
__launch_bounds__(32)
__global__ void sum2_logsoftmax(const float* __restrict__ hidden, // [64,512]
                                const float* __restrict__ xn,     // [64,512]
                                const float* __restrict__ W,      // [1024,16]
                                const float* __restrict__ bias,   // [16]
                                float* __restrict__ logp) {       // [64,16]
  const int b = blockIdx.x;
  const int lane = threadIdx.x & 31;
  const int o = lane & 15;          // lanes 16-31 mirror lanes 0-15
  float s = bias[o];
  for (int i = 0; i < 512; ++i) s += hidden[b * 512 + i] * W[i * 16 + o];
  for (int i = 0; i < 512; ++i) s += xn[b * 512 + i] * W[(512 + i) * 16 + o];
  float mx = s;
#pragma unroll
  for (int d = 1; d < 16; d <<= 1) mx = fmaxf(mx, __shfl_xor(mx, d, 32));
  float e = __expf(s - mx);
  float sm = e;
#pragma unroll
  for (int d = 1; d < 16; d <<= 1) sm += __shfl_xor(sm, d, 32);
  if (lane < 16) logp[b * 16 + lane] = s - mx - __logf(sm);
}

// ---------------------------------------------------------------------------
extern "C" void kernel_launch(void* const* d_in, const int* in_sizes, int n_in,
                              void* d_out, int out_size, void* d_ws, size_t ws_size,
                              hipStream_t stream) {
  const float* x_in   = (const float*)d_in[0];
  const float* Lm     = (const float*)d_in[1];
  // d_in[2] = dropout rate d == 0 -> identity
  const float* cl1_W  = (const float*)d_in[3];
  const float* cl1_b  = (const float*)d_in[4];
  const float* fc1_W  = (const float*)d_in[5];
  const float* fc1_b  = (const float*)d_in[6];
  const float* fc2_W  = (const float*)d_in[7];
  const float* fc2_b  = (const float*)d_in[8];
  const float* fc3_W  = (const float*)d_in[9];
  const float* fc3_b  = (const float*)d_in[10];
  const float* nn1_W  = (const float*)d_in[11];
  const float* nn1_b  = (const float*)d_in[12];
  const float* nn2_W  = (const float*)d_in[13];
  const float* nn2_b  = (const float*)d_in[14];
  const float* sum2_W = (const float*)d_in[15];
  const float* sum2_b = (const float*)d_in[16];

  float* out    = (float*)d_out;
  float* dec    = out;                       // [64, 8192]
  float* hidden = out + (size_t)64 * 8192;   // [64, 512]
  float* logp   = hidden + (size_t)64 * 512; // [64, 16]

  char*  ws  = (char*)d_ws;
  size_t off = 0;
  auto alloc = [&](size_t bytes) -> char* {
    char* p = ws + off;
    off = (off + bytes + 255) & ~(size_t)255;
    return p;
  };
  float* T_all   = (float*)alloc((size_t)5 * VB * 4);  // T0..T4 [V][B]
  bf16*  Tb_all  = (bf16*) alloc((size_t)5 * VB * 2);  // transposed bf16 copies
  float* Ypart   = (float*)alloc((size_t)KSPLIT * VB * 4); // K-split partials
  bf16*  xcp     = (bf16*) alloc((size_t)64 * 16384 * 2);
  bf16*  fc1_Wt  = (bf16*) alloc((size_t)512 * 16384 * 2);
  bf16*  fc2_Wt  = (bf16*) alloc((size_t)256 * 512 * 2);
  bf16*  fc3_Wt  = (bf16*) alloc((size_t)8192 * 256 * 2);
  bf16*  nn1_Wt  = (bf16*) alloc((size_t)1024 * 8192 * 2);
  bf16*  nn2_Wt  = (bf16*) alloc((size_t)512 * 1024 * 2);
  bf16*  hiddenb = (bf16*) alloc((size_t)64 * 512 * 2);
  float* dec_h   = (float*)alloc((size_t)64 * 256 * 4);
  bf16*  dec_hb  = (bf16*) alloc((size_t)64 * 256 * 2);
  float* xn1_f   = (float*)alloc((size_t)64 * 1024 * 4);
  bf16*  xn1_b   = (bf16*) alloc((size_t)64 * 1024 * 2);
  float* xn      = (float*)alloc((size_t)64 * 512 * 4);
  (void)ws_size; (void)in_sizes; (void)n_in; (void)out_size;

  const dim3 tb(32, 8);
  transpose_w_bf16<<<dim3(16384 / 32, 512 / 32), tb, 0, stream>>>(fc1_W, fc1_Wt, 16384, 512);
  transpose_w_bf16<<<dim3(512 / 32, 256 / 32),  tb, 0, stream>>>(fc2_W, fc2_Wt, 512, 256);
  transpose_w_bf16<<<dim3(256 / 32, 8192 / 32), tb, 0, stream>>>(fc3_W, fc3_Wt, 256, 8192);
  transpose_w_bf16<<<dim3(8192 / 32, 1024 / 32),tb, 0, stream>>>(nn1_W, nn1_Wt, 8192, 1024);
  transpose_w_bf16<<<dim3(1024 / 32, 512 / 32), tb, 0, stream>>>(nn2_W, nn2_Wt, 1024, 512);

  float* T0 = T_all;                         bf16* Tb0 = Tb_all;
  prep_x<<<(64 * V_DIM) / 256, 256, 0, stream>>>(x_in, T0, Tb0);

  // Chebyshev recursion: T1 = L@T0 ; Tk = 2*(L@T_{k-1}) - T_{k-2}
  for (int k = 1; k < 5; ++k) {
    float* Tk        = T_all + (size_t)k * VB;
    bf16*  Tbk       = Tb_all + (size_t)k * VB;
    const bf16* Xt   = Tb_all + (size_t)(k - 1) * VB;
    const float* Cp  = (k >= 2) ? (T_all + (size_t)(k - 2) * VB) : nullptr;
    const float alpha = (k >= 2) ? 2.0f : 1.0f;
    cheb_gemm_part<<<dim3(V_DIM / 32, KSPLIT), 64, 0, stream>>>(Lm, Xt, Ypart);
    cheb_combine<<<(int)(VB / 256), 256, 0, stream>>>(Ypart, Cp, alpha, Tk, Tbk);
  }

  cl1_pool<<<1024, dim3(64, 16), 0, stream>>>(T_all, cl1_W, cl1_b, xcp);

  // hidden = relu(xcp @ fc1_W + b) -> d_out + bf16 mirror
  fc_gemm<<<512 / 64, 128, 0, stream>>>(xcp, fc1_Wt, fc1_b, hidden, hiddenb, 16384, 512, 1);
  // dec = relu(hidden @ fc2_W + b) @ fc3_W + b
  fc_gemm<<<256 / 64, 128, 0, stream>>>(hiddenb, fc2_Wt, fc2_b, dec_h, dec_hb, 512, 256, 1);
  fc_gemm<<<8192 / 64, 128, 0, stream>>>(dec_hb, fc3_Wt, fc3_b, dec, nullptr, 256, 8192, 0);
  // nn branch: xn = relu(relu(x @ nn1 + b) @ nn2 + b)   (Tb0 == bf16 x_in)
  fc_gemm<<<1024 / 64, 128, 0, stream>>>(Tb0, nn1_Wt, nn1_b, xn1_f, xn1_b, 8192, 1024, 1);
  fc_gemm<<<512 / 64, 128, 0, stream>>>(xn1_b, nn2_Wt, nn2_b, xn, nullptr, 1024, 512, 1);

  sum2_logsoftmax<<<64, 32, 0, stream>>>(hidden, xn, sum2_W, sum2_b, logp);
}
